// DINOv3ViTAttention_78924319031856
// MI455X (gfx1250) — matrix operand order repaired and checked by reading, not compiled, not run on hardware
//
#include <hip/hip_runtime.h>
#include <hip/hip_bf16.h>
#include <stdint.h>

// ---------------------------------------------------------------------------
// DINOv3 ViT attention for MI455X (gfx1250, wave32, WMMA).
// Compute-bound (~118 GFLOP vs ~100MB traffic) -> everything heavy runs on
// v_wmma_f32_16x16x32_bf16 with f32 accumulation.
// GEMM: 64x64 tile per wave (4x4 WMMA frags) -> 1:1 load:WMMA in the K loop.
// ---------------------------------------------------------------------------

#define B_      8
#define N_HID   1029
#define N_DET   100
#define SEQ     1129          // N_DET + N_HID
#define EMB     1024
#define NHEADS  16
#define HDIM    64
#define PLEN    1024
#define RSTART  105           // SEQ - PLEN
#define S_PAD   1216          // >= 100 + 17*64 (q-hidden tiles), mult of 64
#define KV_PAD  1152          // 18 * 64 rows of K/V written; 36 * 32 kv tiles
#define SCALE_F 0.125f        // 64^-0.5

typedef unsigned short u16;
typedef __attribute__((ext_vector_type(16))) __bf16        v16bf;
typedef __attribute__((ext_vector_type(8)))  float         v8f;
typedef __attribute__((ext_vector_type(4)))  unsigned int  v4u;

union FragBF { v16bf v; v4u u[2]; };

__device__ __forceinline__ u16 f2bf(float x) {
    unsigned u = __float_as_uint(x);
    unsigned r = u + 0x7FFFu + ((u >> 16) & 1u);   // round-to-nearest-even
    return (u16)(r >> 16);
}
__device__ __forceinline__ float bf2f(u16 h) {
    return __uint_as_float(((unsigned)h) << 16);
}
__device__ __forceinline__ v8f v8zero() {
    v8f z;
#pragma unroll
    for (int i = 0; i < 8; ++i) z[i] = 0.f;
    return z;
}

// ---------------------------------------------------------------------------
// Pack x = concat(det_tokens, hidden_states) -> bf16, zero-padded to S_PAD.
// ---------------------------------------------------------------------------
__global__ void pack_x_kernel(const float* __restrict__ det,
                              const float* __restrict__ hid,
                              u16* __restrict__ X)
{
    long long t = (long long)blockIdx.x * blockDim.x + threadIdx.x;
    const long long total = (long long)B_ * S_PAD * EMB;
    if (t >= total) return;
    int e = (int)(t & (EMB - 1));
    long long rs = t >> 10;
    int b = (int)(rs / S_PAD);
    int s = (int)(rs - (long long)b * S_PAD);
    float v = 0.f;
    if (s < N_DET)      v = det[((long long)b * N_DET + s) * EMB + e];
    else if (s < SEQ)   v = hid[((long long)b * N_HID + (s - N_DET)) * EMB + e];
    X[t] = f2bf(v);
}

__global__ void cvt_f32_bf16_kernel(const float* __restrict__ src,
                                    u16* __restrict__ dst, int n)
{
    int t = blockIdx.x * blockDim.x + threadIdx.x;
    if (t < n) dst[t] = f2bf(src[t]);
}

// ---------------------------------------------------------------------------
// bf16 GEMM: C[m,n] = sum_k A[m,k] * W[n,k] (+bias[n]).  One wave per 64x64
// output tile: 4x4 WMMA fragment grid, K step 32 -> 16 b128 loads feed
// 16 v_wmma per K step (1:1), 128 f32 accumulator VGPRs.
// mode 0: bf16 row-major C            (ldc = row stride)
// mode 1: bf16 transposed C[n*ldc+m]  (for V^T)
// mode 2: f32 row-major C + bias      (final output, row-guarded)
// strideAb/strideCb are per-batch element strides (grid.z = batch).
// ---------------------------------------------------------------------------
__global__ __launch_bounds__(32) void gemm_bf16_kernel(
    const u16* __restrict__ A, long long strideAb, int lda,
    const u16* __restrict__ W, int ldw,
    void* __restrict__ Cv, long long strideCb, int ldc,
    const float* __restrict__ bias,
    int Mvalid, int Kdim, int mode)
{
    const int lane = threadIdx.x;
    const int half = lane >> 4;
    const int l16  = lane & 15;
    const int mbase = blockIdx.x * 64;
    const int nbase = blockIdx.y * 64;
    A += (long long)blockIdx.z * strideAb;

    const u16* arow[4];
    const u16* wrow[4];
#pragma unroll
    for (int i = 0; i < 4; ++i) {
        arow[i] = A + (long long)(mbase + i * 16 + l16) * lda;
        wrow[i] = W + (long long)(nbase + i * 16 + l16) * ldw;
    }
    const int ca = half * 8;    // A-frag: 8-half chunks split by lane-half
    const int cb = half * 8;    // B-frag: 8-half chunks split by lane-half

    v8f acc[4][4];
#pragma unroll
    for (int i = 0; i < 4; ++i)
#pragma unroll
        for (int j = 0; j < 4; ++j) acc[i][j] = v8zero();

    for (int kb = 0; kb < Kdim; kb += 32) {
        FragBF a[4], b[4];
#pragma unroll
        for (int i = 0; i < 4; ++i) {
            a[i].u[0] = *(const v4u*)(arow[i] + kb + ca);
            a[i].u[1] = *(const v4u*)(arow[i] + kb + 16 + ca);
        }
#pragma unroll
        for (int j = 0; j < 4; ++j) {
            b[j].u[0] = *(const v4u*)(wrow[j] + kb + cb);
            b[j].u[1] = *(const v4u*)(wrow[j] + kb + cb + 16);
        }
#pragma unroll
        for (int i = 0; i < 4; ++i)
#pragma unroll
            for (int j = 0; j < 4; ++j)
                acc[i][j] = __builtin_amdgcn_wmma_f32_16x16x32_bf16(
                    false, a[i].v, false, b[j].v, (short)0, acc[i][j], false, false);
    }

#pragma unroll
    for (int i = 0; i < 4; ++i) {
#pragma unroll
        for (int j = 0; j < 4; ++j) {
#pragma unroll
            for (int r = 0; r < 8; ++r) {
                int m = mbase + i * 16 + r + half * 8;   // C layout: lanes 16-31 -> M+8
                int n = nbase + j * 16 + l16;
                if (m >= Mvalid) continue;
                float vv = acc[i][j][r] + (bias ? bias[n] : 0.f);
                if (mode == 2) {
                    ((float*)Cv + (long long)blockIdx.z * strideCb)[(long long)m * ldc + n] = vv;
                } else if (mode == 1) {
                    ((u16*)Cv + (long long)blockIdx.z * strideCb)[(long long)n * ldc + m] = f2bf(vv);
                } else {
                    ((u16*)Cv + (long long)blockIdx.z * strideCb)[(long long)m * ldc + n] = f2bf(vv);
                }
            }
        }
    }
}

// ---------------------------------------------------------------------------
// RoPE on Q and K for positions s in [RSTART, SEQ).  rotate_half semantics:
// out[d]    = x[d]   *cos[d]    - x[d+32]*sin[d]
// out[d+32] = x[d+32]*cos[d+32] + x[d]   *sin[d+32]
// ---------------------------------------------------------------------------
__global__ void rope_kernel(u16* __restrict__ Q, u16* __restrict__ Kb,
                            const float* __restrict__ cosb,
                            const float* __restrict__ sinb)
{
    int t = blockIdx.x * blockDim.x + threadIdx.x;  // B*P*H*32 threads
    int d = t & 31;
    int h = (t >> 5) & (NHEADS - 1);
    int p = (t >> 9) & (PLEN - 1);
    int b = t >> 19;
    if (b >= B_) return;
    long long base = ((long long)b * S_PAD + RSTART + p) * EMB + h * HDIM + d;
    float c1 = cosb[p * HDIM + d],      s1 = sinb[p * HDIM + d];
    float c2 = cosb[p * HDIM + 32 + d], s2 = sinb[p * HDIM + 32 + d];
    float q1 = bf2f(Q[base]), q2 = bf2f(Q[base + 32]);
    Q[base]      = f2bf(q1 * c1 - q2 * s1);
    Q[base + 32] = f2bf(q2 * c2 + q1 * s2);
    float k1 = bf2f(Kb[base]), k2 = bf2f(Kb[base + 32]);
    Kb[base]      = f2bf(k1 * c1 - k2 * s1);
    Kb[base + 32] = f2bf(k2 * c2 + k1 * s2);
}

// ---------------------------------------------------------------------------
// Flash attention: grid (ceil(S/64), NHEADS, B), 4 waves/block, each wave owns
// 16 query rows.  Per 32-key tile: QK^T via 4 WMMAs, online softmax with
// shfl-based row reductions, P transposed C->A layout through a private LDS
// tile, then P@V via 4 WMMAs.  Keys >= SEQ masked to -inf.
// ---------------------------------------------------------------------------
__global__ __launch_bounds__(128) void attn_kernel(
    const u16* __restrict__ Q, const u16* __restrict__ Kb,
    const u16* __restrict__ VT, u16* __restrict__ AO)
{
    __shared__ alignas(16) u16 pbuf[4][16][32];   // per-wave P staging tile
    const int wave = threadIdx.x >> 5;
    const int lane = threadIdx.x & 31;
    const int half = lane >> 4;
    const int l16  = lane & 15;
    const int b = blockIdx.z;
    const int h = blockIdx.y;
    const int qbase = blockIdx.x * 64 + wave * 16;
    const int ca = half * 8;
    const int cb = half * 8;

    // Q A-fragments: rows qbase+l16, head dims [0,32) and [32,64)
    long long qoff = ((long long)b * S_PAD + qbase + l16) * EMB + h * HDIM;
    FragBF aq0, aq1;
    aq0.u[0] = *(const v4u*)(Q + qoff + ca);
    aq0.u[1] = *(const v4u*)(Q + qoff + 16 + ca);
    aq1.u[0] = *(const v4u*)(Q + qoff + 32 + ca);
    aq1.u[1] = *(const v4u*)(Q + qoff + 48 + ca);

    float mrow[8], lrow[8];
    v8f acc0 = v8zero(), acc1 = v8zero(), acc2 = v8zero(), acc3 = v8zero();
#pragma unroll
    for (int r = 0; r < 8; ++r) { mrow[r] = -1e30f; lrow[r] = 0.f; }

    for (int kb = 0; kb < KV_PAD; kb += 32) {
        // K B-fragments: keys kb..kb+31 (two 16-key groups), dims 0..63
        long long k0off = ((long long)b * S_PAD + kb + l16) * EMB + h * HDIM;
        long long k1off = ((long long)b * S_PAD + kb + 16 + l16) * EMB + h * HDIM;
        FragBF bk00, bk01, bk10, bk11;
        bk00.u[0] = *(const v4u*)(Kb + k0off + cb);
        bk00.u[1] = *(const v4u*)(Kb + k0off + cb + 16);
        bk01.u[0] = *(const v4u*)(Kb + k0off + 32 + cb);
        bk01.u[1] = *(const v4u*)(Kb + k0off + 32 + cb + 16);
        bk10.u[0] = *(const v4u*)(Kb + k1off + cb);
        bk10.u[1] = *(const v4u*)(Kb + k1off + cb + 16);
        bk11.u[0] = *(const v4u*)(Kb + k1off + 32 + cb);
        bk11.u[1] = *(const v4u*)(Kb + k1off + 32 + cb + 16);

        v8f s0 = v8zero(), s1 = v8zero();
        s0 = __builtin_amdgcn_wmma_f32_16x16x32_bf16(false, aq0.v, false, bk00.v, (short)0, s0, false, false);
        s0 = __builtin_amdgcn_wmma_f32_16x16x32_bf16(false, aq1.v, false, bk01.v, (short)0, s0, false, false);
        s1 = __builtin_amdgcn_wmma_f32_16x16x32_bf16(false, aq0.v, false, bk10.v, (short)0, s1, false, false);
        s1 = __builtin_amdgcn_wmma_f32_16x16x32_bf16(false, aq1.v, false, bk11.v, (short)0, s1, false, false);

        const bool ok0 = (kb + l16) < SEQ;
        const bool ok1 = (kb + 16 + l16) < SEQ;

        float p0[8], p1[8];
#pragma unroll
        for (int r = 0; r < 8; ++r) {
            float f0 = ok0 ? s0[r] * SCALE_F : -1e30f;
            float f1 = ok1 ? s1[r] * SCALE_F : -1e30f;
            float tm = fmaxf(f0, f1);
            tm = fmaxf(tm, __shfl_xor(tm, 1, 32));
            tm = fmaxf(tm, __shfl_xor(tm, 2, 32));
            tm = fmaxf(tm, __shfl_xor(tm, 4, 32));
            tm = fmaxf(tm, __shfl_xor(tm, 8, 32));
            float mn = fmaxf(mrow[r], tm);
            float sc = __expf(mrow[r] - mn);
            mrow[r] = mn;
            p0[r] = ok0 ? __expf(f0 - mn) : 0.f;
            p1[r] = ok1 ? __expf(f1 - mn) : 0.f;
            lrow[r] = lrow[r] * sc + p0[r] + p1[r];
            acc0[r] *= sc; acc1[r] *= sc; acc2[r] *= sc; acc3[r] *= sc;
        }

        // Stage P (C layout) into LDS, read back as A layout.  Same-wave LDS
        // ops complete in order; asm fence stops compiler reordering.
#pragma unroll
        for (int r = 0; r < 8; ++r) {
            pbuf[wave][r + half * 8][l16]      = f2bf(p0[r]);
            pbuf[wave][r + half * 8][16 + l16] = f2bf(p1[r]);
        }
        asm volatile("" ::: "memory");
        FragBF ap;
        ap.u[0] = *(const v4u*)&pbuf[wave][l16][ca];
        ap.u[1] = *(const v4u*)&pbuf[wave][l16][16 + ca];
        asm volatile("" ::: "memory");

        // P @ V via V^T (rows = dims, contiguous keys)
        long long vbase = ((long long)b * EMB + h * HDIM) * S_PAD + kb + cb;
        FragBF bv;
#define PV_STEP(dg, accv)                                                     \
        {                                                                     \
            long long vo = vbase + (long long)((dg) * 16 + l16) * S_PAD;      \
            bv.u[0] = *(const v4u*)(VT + vo);                                 \
            bv.u[1] = *(const v4u*)(VT + vo + 16);                            \
            accv = __builtin_amdgcn_wmma_f32_16x16x32_bf16(                   \
                false, ap.v, false, bv.v, (short)0, accv, false, false);      \
        }
        PV_STEP(0, acc0)
        PV_STEP(1, acc1)
        PV_STEP(2, acc2)
        PV_STEP(3, acc3)
#undef PV_STEP
    }

#pragma unroll
    for (int r = 0; r < 8; ++r) {
        float l = lrow[r];
        l += __shfl_xor(l, 1, 32);
        l += __shfl_xor(l, 2, 32);
        l += __shfl_xor(l, 4, 32);
        l += __shfl_xor(l, 8, 32);
        float inv = 1.f / l;
        long long obase = ((long long)b * S_PAD + qbase + r + half * 8) * EMB
                        + h * HDIM + l16;
        AO[obase]      = f2bf(acc0[r] * inv);
        AO[obase + 16] = f2bf(acc1[r] * inv);
        AO[obase + 32] = f2bf(acc2[r] * inv);
        AO[obase + 48] = f2bf(acc3[r] * inv);
    }
}

// ---------------------------------------------------------------------------
// Host orchestration.  Workspace (~110 MB): X, Q, K, V^T, AO bf16 buffers of
// B*S_PAD*EMB + 5 bf16 weight copies.
// ---------------------------------------------------------------------------
extern "C" void kernel_launch(void* const* d_in, const int* in_sizes, int n_in,
                              void* d_out, int out_size, void* d_ws, size_t ws_size,
                              hipStream_t stream)
{
    const float* hid  = (const float*)d_in[0];
    const float* det  = (const float*)d_in[1];
    const float* cosb = (const float*)d_in[2];
    const float* sinb = (const float*)d_in[3];
    const float* q_w  = (const float*)d_in[4];
    const float* q_b  = (const float*)d_in[5];
    const float* k_w  = (const float*)d_in[6];
    const float* v_w  = (const float*)d_in[7];
    const float* v_b  = (const float*)d_in[8];
    const float* dq_w = (const float*)d_in[9];
    const float* dq_b = (const float*)d_in[10];
    const float* o_w  = (const float*)d_in[11];
    const float* o_b  = (const float*)d_in[12];
    (void)in_sizes; (void)n_in; (void)out_size; (void)ws_size;

    const size_t nX = (size_t)B_ * S_PAD * EMB;
    u16* X   = (u16*)d_ws;
    u16* Qb  = X   + nX;
    u16* Kb  = Qb  + nX;
    u16* VT  = Kb  + nX;
    u16* AO  = VT  + nX;
    u16* Wq  = AO  + nX;
    u16* Wk  = Wq  + (size_t)EMB * EMB;
    u16* Wv  = Wk  + (size_t)EMB * EMB;
    u16* Wdq = Wv  + (size_t)EMB * EMB;
    u16* Wo  = Wdq + (size_t)EMB * EMB;

    {   // pack + zero-pad x
        long long total = (long long)B_ * S_PAD * EMB;
        pack_x_kernel<<<(unsigned)((total + 255) / 256), 256, 0, stream>>>(det, hid, X);
    }
    {   // weights -> bf16
        int wn = EMB * EMB, wb = (wn + 255) / 256;
        cvt_f32_bf16_kernel<<<wb, 256, 0, stream>>>(q_w,  Wq,  wn);
        cvt_f32_bf16_kernel<<<wb, 256, 0, stream>>>(k_w,  Wk,  wn);
        cvt_f32_bf16_kernel<<<wb, 256, 0, stream>>>(v_w,  Wv,  wn);
        cvt_f32_bf16_kernel<<<wb, 256, 0, stream>>>(dq_w, Wdq, wn);
        cvt_f32_bf16_kernel<<<wb, 256, 0, stream>>>(o_w,  Wo,  wn);
    }

    const long long sAB = (long long)S_PAD * EMB;

    // K = x @ k_w^T              (full KV_PAD rows so pad rows are defined)
    gemm_bf16_kernel<<<dim3(18, 16, B_), 32, 0, stream>>>(
        X, sAB, EMB, Wk, EMB, Kb, sAB, EMB, nullptr, KV_PAD, EMB, 0);
    // V^T = (x @ v_w^T + v_b)^T
    gemm_bf16_kernel<<<dim3(18, 16, B_), 32, 0, stream>>>(
        X, sAB, EMB, Wv, EMB, VT, (long long)EMB * S_PAD, S_PAD, v_b, KV_PAD, EMB, 1);
    // Q (det rows, dq_w)
    gemm_bf16_kernel<<<dim3(2, 16, B_), 32, 0, stream>>>(
        X, sAB, EMB, Wdq, EMB, Qb, sAB, EMB, dq_b, N_DET, EMB, 0);
    // Q (hidden rows, q_w)
    gemm_bf16_kernel<<<dim3(17, 16, B_), 32, 0, stream>>>(
        X + (size_t)N_DET * EMB, sAB, EMB, Wq, EMB,
        Qb + (size_t)N_DET * EMB, sAB, EMB, q_b, N_HID, EMB, 0);

    {   // RoPE on Q and K
        int total = B_ * PLEN * NHEADS * 32;
        rope_kernel<<<total / 256, 256, 0, stream>>>(Qb, Kb, cosb, sinb);
    }

    // Flash attention -> AO (bf16, row-major [b][s][e])
    attn_kernel<<<dim3((KV_PAD / 64), NHEADS, B_), 128, 0, stream>>>(Qb, Kb, VT, AO);

    // out = AO @ o_w^T + o_b   (f32, row-guarded to SEQ)
    gemm_bf16_kernel<<<dim3(18, 16, B_), 32, 0, stream>>>(
        AO, sAB, EMB, Wo, EMB, d_out, (long long)SEQ * EMB, EMB, o_b, SEQ, EMB, 2);
}
